// VQ2Linear_70978629534416
// MI455X (gfx1250) — matrix operand order repaired and checked by reading, not compiled, not run on hardware
//
#include <hip/hip_runtime.h>
#include <hip/hip_bf16.h>

// VQ-VAE quantization for MI455X (gfx1250, wave32, WMMA).
// GEMM z @ emb^T fused with argmin, bf16x3 split for ~f32 accuracy on the
// v_wmma_f32_16x16x32_bf16 pipe. Codebook (8MB) is L2-resident; double-
// buffered LDS staging via async global->LDS (ASYNCcnt) when available.

#define B_ROWS   32768
#define N_E      8192
#define E_DIM    256
#define BETA     0.25f

#define NCH        32      // codebook entries staged in LDS per iteration
#define LDS_STRIDE 272     // ushorts per staged row: 256 + 16 pad (544B, 32B-aligned)
#define KSTEPS     8       // 256 / 32

#if defined(__has_builtin)
#  if __has_builtin(__builtin_amdgcn_global_load_async_to_lds_b128) && \
      __has_builtin(__builtin_amdgcn_s_wait_asynccnt)
#    define USE_ASYNC_LDS 1
#  endif
#endif
#ifndef USE_ASYNC_LDS
#  define USE_ASYNC_LDS 0
#endif

typedef __attribute__((ext_vector_type(16))) __bf16         v16bf;
typedef __attribute__((ext_vector_type(16))) unsigned short v16u16;
typedef __attribute__((ext_vector_type(8)))  unsigned short v8u16;
typedef __attribute__((ext_vector_type(8)))  float          v8f;
typedef int v4i __attribute__((vector_size(16)));   // matches builtin param type

union BF16x16 { v16u16 u; v16bf b; };

__device__ __forceinline__ unsigned short bf16_trunc_bits(float x) {
    return (unsigned short)(__float_as_uint(x) >> 16);
}

#if USE_ASYNC_LDS
__device__ __forceinline__
__attribute__((address_space(3))) v4i* lds_v4i(void* p) {
    return (__attribute__((address_space(3))) v4i*)p;
}
__device__ __forceinline__
__attribute__((address_space(1))) v4i* glob_v4i(const void* p) {
    return (__attribute__((address_space(1))) v4i*)((void*)p);
}
#endif

#define WMMA_BF16(A, B, C) \
    __builtin_amdgcn_wmma_f32_16x16x32_bf16(false, (A), false, (B), (short)0, (C), false, false)

// ---------------------------------------------------------------------------
// Prep: split codebook into hi/lo bf16 planes, compute ||e||^2, zero loss slot.
// One wave32 per codebook row (8 f32 per lane).
// ---------------------------------------------------------------------------
__global__ __launch_bounds__(256)
void vq_prep(const float* __restrict__ emb,
             unsigned short* __restrict__ ehi,
             unsigned short* __restrict__ elo,
             float* __restrict__ enorm,
             float* __restrict__ loss_slot) {
    if (blockIdx.x == 0 && threadIdx.x == 0) *loss_slot = 0.0f;

    const int gid  = blockIdx.x * blockDim.x + threadIdx.x;
    const int row  = gid >> 5;
    const int lane = gid & 31;
    if (row >= N_E) return;

    const float* src = emb + (size_t)row * E_DIM + lane * 8;
    v8u16 h, l;
    float s = 0.0f;
#pragma unroll
    for (int i = 0; i < 8; ++i) {
        const float x = src[i];
        s += x * x;
        const unsigned ux = __float_as_uint(x);
        const float hf = __uint_as_float(ux & 0xFFFF0000u);
        h[i] = (unsigned short)(ux >> 16);
        l[i] = bf16_trunc_bits(x - hf);
    }
    *(v8u16*)(ehi + (size_t)row * E_DIM + lane * 8) = h;
    *(v8u16*)(elo + (size_t)row * E_DIM + lane * 8) = l;

#pragma unroll
    for (int m = 16; m >= 1; m >>= 1) s += __shfl_xor(s, m, 32);
    if (lane == 0) enorm[row] = s;
}

// ---------------------------------------------------------------------------
// Stage NCH codebook rows (hi + lo planes) into one LDS buffer.
// Async path: 16B per lane per op, tracked by ASYNCcnt (no VGPR data).
// ---------------------------------------------------------------------------
__device__ __forceinline__
void stage_chunk(int tid, int nb,
                 const unsigned short* __restrict__ ehi,
                 const unsigned short* __restrict__ elo,
                 unsigned short* dHi, unsigned short* dLo) {
#pragma unroll
    for (int k = 0; k < 4; ++k) {
        const int idx = tid + k * 256;        // 1024 segments of 16B per plane
        const int r   = idx >> 5;
        const int seg = idx & 31;
        const size_t g = (size_t)(nb + r) * E_DIM + seg * 8;
        const int    o = r * LDS_STRIDE + seg * 8;
#if USE_ASYNC_LDS
        __builtin_amdgcn_global_load_async_to_lds_b128(
            glob_v4i(ehi + g), lds_v4i(dHi + o), 0, 0);
        __builtin_amdgcn_global_load_async_to_lds_b128(
            glob_v4i(elo + g), lds_v4i(dLo + o), 0, 0);
#else
        *(v8u16*)(dHi + o) = *(const v8u16*)(ehi + g);
        *(v8u16*)(dLo + o) = *(const v8u16*)(elo + g);
#endif
    }
}

// ---------------------------------------------------------------------------
// Main: 128 z-rows per workgroup (8 waves x 16 rows). Fused GEMM + argmin +
// gather + loss. A fragments (hi/lo, all K) live in registers per wave.
// ---------------------------------------------------------------------------
__global__ __launch_bounds__(256)
void vq_main(const float* __restrict__ z,
             const float* __restrict__ emb,
             const unsigned short* __restrict__ ehi,
             const unsigned short* __restrict__ elo,
             const float* __restrict__ enorm,
             float* __restrict__ out,
             float* __restrict__ loss_slot) {
    __shared__ __align__(32) unsigned short sHi[2][NCH * LDS_STRIDE];
    __shared__ __align__(32) unsigned short sLo[2][NCH * LDS_STRIDE];

    const int tid  = threadIdx.x;
    const int wave = tid >> 5;
    const int lane = tid & 31;
    const int half = lane >> 4;   // A/B K-half selector per WMMA lane layout
    const int ln   = lane & 15;

    const int rowBase = blockIdx.x * 128 + wave * 16;

    // ---- load + split this wave's 16 z rows into WMMA A fragments ----
    // A k order used here: lane (half==0): K = k0..k0+15, k0=ks*32
    //                  lane (half==1): same but k0 += 16.
    v16bf Ahi[KSTEPS], Alo[KSTEPS];
    {
        const float* zrow = z + (size_t)(rowBase + ln) * E_DIM;
#pragma unroll
        for (int ks = 0; ks < KSTEPS; ++ks) {
            const int k0 = ks * 32 + half * 16;
            BF16x16 ah, al;
#pragma unroll
            for (int i = 0; i < 8; ++i) {
                const float x = zrow[k0 + i];
                const unsigned ux = __float_as_uint(x);
                ah.u[i] = (unsigned short)(ux >> 16);
                al.u[i] = bf16_trunc_bits(x - __uint_as_float(ux & 0xFFFF0000u));
                const float y = zrow[k0 + 8 + i];
                const unsigned uy = __float_as_uint(y);
                ah.u[8 + i] = (unsigned short)(uy >> 16);
                al.u[8 + i] = bf16_trunc_bits(y - __uint_as_float(uy & 0xFFFF0000u));
            }
            Ahi[ks] = ah.b;
            Alo[ks] = al.b;
        }
    }

    // Running argmin state, matching the 16x16 f32 C layout:
    // lane half h, VGPR r  <->  row M = 8*h + r, col N = ln (within sub-chunk).
    float bestd[8];
    int   bestn[8];
#pragma unroll
    for (int r = 0; r < 8; ++r) { bestd[r] = 3.4e38f; bestn[r] = 0x7fffffff; }

    // Prologue: stage chunk 0 into buffer 0.
    stage_chunk(tid, 0, ehi, elo, sHi[0], sLo[0]);
#if USE_ASYNC_LDS
    __builtin_amdgcn_s_wait_asynccnt(0);
#endif
    __syncthreads();

    int buf = 0;
    for (int nb = 0; nb < N_E; nb += NCH) {
        // Kick off the async stage of the next chunk into the spare buffer.
        const int nxt = nb + NCH;
        if (nxt < N_E)
            stage_chunk(tid, nxt, ehi, elo, sHi[buf ^ 1], sLo[buf ^ 1]);

        const unsigned short* hB = sHi[buf];
        const unsigned short* lB = sLo[buf];
        const float en0 = enorm[nb + ln];
        const float en1 = enorm[nb + 16 + ln];

        // B k order used here (same as A): lane = column (ln), contiguous K-run of 16 selected
        // by lane half -> one 32B LDS read per fragment. Sub-chunk 1 sits 16
        // rows further into the staged tile.
        const unsigned short* bh0 = hB + ln * LDS_STRIDE + half * 16;
        const unsigned short* bl0 = lB + ln * LDS_STRIDE + half * 16;
        const unsigned short* bh1 = bh0 + 16 * LDS_STRIDE;
        const unsigned short* bl1 = bl0 + 16 * LDS_STRIDE;

        v8f acc0 = {}, acc1 = {};
        BF16x16 Bh0, Bl0, Bh1, Bl1, Ph0, Pl0, Ph1, Pl1;
        Bh0.u = *(const v16u16*)(bh0);
        Bl0.u = *(const v16u16*)(bl0);
        Bh1.u = *(const v16u16*)(bh1);
        Bl1.u = *(const v16u16*)(bl1);
#pragma unroll
        for (int ks = 0; ks < KSTEPS; ++ks) {
            // Prefetch next kstep's B fragments while this kstep's WMMAs run.
            if (ks + 1 < KSTEPS) {
                Ph0.u = *(const v16u16*)(bh0 + (ks + 1) * 32);
                Pl0.u = *(const v16u16*)(bl0 + (ks + 1) * 32);
                Ph1.u = *(const v16u16*)(bh1 + (ks + 1) * 32);
                Pl1.u = *(const v16u16*)(bl1 + (ks + 1) * 32);
            }
            // bf16x3 split, two independent accumulator chains (sub0 / sub1).
            acc0 = WMMA_BF16(Ahi[ks], Bh0.b, acc0);
            acc1 = WMMA_BF16(Ahi[ks], Bh1.b, acc1);
            acc0 = WMMA_BF16(Alo[ks], Bh0.b, acc0);
            acc1 = WMMA_BF16(Alo[ks], Bh1.b, acc1);
            acc0 = WMMA_BF16(Ahi[ks], Bl0.b, acc0);
            acc1 = WMMA_BF16(Ahi[ks], Bl1.b, acc1);
            Bh0 = Ph0; Bl0 = Pl0; Bh1 = Ph1; Bl1 = Pl1;
        }

        // score = ||e||^2 - 2 z.e  (||z||^2 constant per row: irrelevant)
        const int n0 = nb + ln;
        const int n1 = nb + 16 + ln;
#pragma unroll
        for (int r = 0; r < 8; ++r) {
            const float d0 = en0 - 2.0f * acc0[r];
            if (d0 < bestd[r] || (d0 == bestd[r] && n0 < bestn[r])) {
                bestd[r] = d0; bestn[r] = n0;
            }
            const float d1 = en1 - 2.0f * acc1[r];
            if (d1 < bestd[r] || (d1 == bestd[r] && n1 < bestn[r])) {
                bestd[r] = d1; bestn[r] = n1;
            }
        }

#if USE_ASYNC_LDS
        __builtin_amdgcn_s_wait_asynccnt(0);   // my async stage done
#endif
        __syncthreads();                        // everyone done; buffers safe
        buf ^= 1;
    }

    // Argmin reduction across the 16 lanes of each half (xor masks 1..8 stay
    // inside a half on wave32).
#pragma unroll
    for (int m = 8; m >= 1; m >>= 1) {
#pragma unroll
        for (int r = 0; r < 8; ++r) {
            const float od = __shfl_xor(bestd[r], m, 32);
            const int   on = __shfl_xor(bestn[r], m, 32);
            if (od < bestd[r] || (od == bestd[r] && on < bestn[r])) {
                bestd[r] = od; bestn[r] = on;
            }
        }
    }

    // Gather z_q = emb[bestn], write output, accumulate loss partial.
    float lsum = 0.0f;
#pragma unroll
    for (int i = 0; i < 16; ++i) {
        const int srcLane = (i >> 3) * 16;              // half holding row i
        const int n_i = __shfl(bestn[i & 7], srcLane, 32);
        const float* e  = emb + (size_t)n_i * E_DIM;
        const float* zr = z   + (size_t)(rowBase + i) * E_DIM;
        float*       o  = out + (size_t)(rowBase + i) * E_DIM;
#pragma unroll
        for (int j = 0; j < 8; ++j) {
            const float ev = e[lane * 8 + j];
            const float zv = zr[lane * 8 + j];
            o[lane * 8 + j] = ev;                       // z_q_st == z_q numerically
            const float df = ev - zv;
            lsum += df * df;
        }
    }
#pragma unroll
    for (int m = 16; m >= 1; m >>= 1) lsum += __shfl_xor(lsum, m, 32);
    if (lane == 0) {
        // loss = (1+BETA) * mean((z_q - z)^2)
        atomicAdd(loss_slot,
                  lsum * ((1.0f + BETA) / (float)((size_t)B_ROWS * E_DIM)));
    }
}

// ---------------------------------------------------------------------------
extern "C" void kernel_launch(void* const* d_in, const int* in_sizes, int n_in,
                              void* d_out, int out_size, void* d_ws, size_t ws_size,
                              hipStream_t stream) {
    (void)in_sizes; (void)n_in; (void)ws_size;

    const float* z   = (const float*)d_in[0];
    const float* emb = (const float*)d_in[1];
    float* out       = (float*)d_out;
    float* loss_slot = out + (size_t)out_size - 1;   // tuple tail: scalar loss

    // Workspace layout: emb_hi (4MB) | emb_lo (4MB) | enorm (32KB)
    unsigned short* ehi = (unsigned short*)d_ws;
    unsigned short* elo = ehi + (size_t)N_E * E_DIM;
    float*          enr = (float*)(elo + (size_t)N_E * E_DIM);

    vq_prep<<<N_E / 8, 256, 0, stream>>>(emb, ehi, elo, enr, loss_slot);
    vq_main<<<B_ROWS / 128, 256, 0, stream>>>(z, emb, ehi, elo, enr, out, loss_slot);
}
